// SimpleNetwork_51608327029023
// MI455X (gfx1250) — matrix operand order repaired and checked by reading, not compiled, not run on hardware
//
#include <hip/hip_runtime.h>

// ---------------------------------------------------------------------------
// Sizes / constants
// ---------------------------------------------------------------------------
#define NN    8192
#define GG    16
#define NPG   512
#define DEG   16
#define C_IN  8
#define MUL   32
#define NSC   96            // 3*MUL
#define KTP   800           // 100*C_IN  (tensor-product GEMM K)
#define NOUT  160           // NSC + MUL + MUL

#define RS8    0.35355339f   // 1/sqrt(8)
#define RS32   0.17677670f   // 1/sqrt(32)
#define RS96   0.10206207f   // 1/sqrt(96)
#define RS3    0.57735027f
#define RS5    0.44721360f
#define RS16   0.25f         // 1/sqrt(NUM_NEI)
#define RS512  0.04419417f   // 1/sqrt(512)

typedef __attribute__((ext_vector_type(16))) _Float16 v16h;
typedef __attribute__((ext_vector_type(8)))  _Float16 v8h;
typedef __attribute__((ext_vector_type(8)))  float    v8f;

__device__ __forceinline__ float f_silu(float v)    { return v / (1.f + __expf(-v)); }
__device__ __forceinline__ float f_sigmoid(float v) { return 1.f / (1.f + __expf(-v)); }

__device__ __forceinline__ v16h mk16(v8h a, v8h b) {
  v16h r;
#pragma unroll
  for (int i = 0; i < 8; ++i) { r[i] = a[i]; r[i + 8] = b[i]; }
  return r;
}

__device__ __forceinline__ v8f wmma16(v16h a, v16h b, v8f c) {
  return __builtin_amdgcn_wmma_f32_16x16x32_f16(false, a, false, b, (short)0, c,
                                                false, false);
}

// CDNA5 async global->LDS copy (GVS mode: saddr base + 32-bit per-lane offset).
// vdst = per-lane LDS byte address, tracked with ASYNCcnt.
__device__ __forceinline__ void async_ld_b128(unsigned lds_addr, unsigned gv_off,
                                              unsigned long long base) {
  asm volatile("global_load_async_to_lds_b128 %0, %1, %2 offset:0"
               :: "v"(lds_addr), "v"(gv_off), "s"(base) : "memory");
}

// ---------------------------------------------------------------------------
// Kernel 0: weight repack (fold all scale factors into f16 B panels) + xs
// ---------------------------------------------------------------------------
#define R0 65536            // xs_h / xh            (N*8)
#define R1 (R0 + 128000)    // Bt1  [160][800]
#define R2 (R1 + 12288)     // Bt2  [96][128]
#define R3 (R2 + 12288)     // Bs   [96][128]
#define R4 (R3 + 5120)      // BtLin[5][32][32]
__global__ void k_prep(const float* __restrict__ x,    const float* __restrict__ lin1a,
                       const float* __restrict__ Wf2a, const float* __restrict__ Wf2b,
                       const float* __restrict__ lin2a0, const float* __restrict__ sc1,
                       const float* __restrict__ lin2a1, const float* __restrict__ lin2a2,
                       const float* __restrict__ lin1b0, const float* __restrict__ lin1b1,
                       const float* __restrict__ lin1b2,
                       _Float16* xs_h, _Float16* xh, _Float16* Bt1, _Float16* Bt2,
                       _Float16* Bs, _Float16* BtLin) {
  int idx = blockIdx.x * 256 + threadIdx.x;
  if (idx >= R4) return;
  if (idx < R0) {                               // xs = x@lin1a * rs8, xh = x
    int n = idx >> 3, c2 = idx & 7;
    float s = 0.f;
#pragma unroll
    for (int c = 0; c < 8; ++c) s += x[n * 8 + c] * lin1a[c * 8 + c2];
    xs_h[idx] = (_Float16)(s * RS8);
    xh[idx]   = (_Float16)x[idx];
  } else if (idx < R1) {                        // tensor-product weights, k-major->Bt
    int t = idx - R0, col = t / KTP, k = t - col * KTP;
    int j = k >> 3, c = k & 7, base, u;
    if (col < NSC)            { base = 0;    u = col; }
    else if (col < NSC + MUL) { base = 768;  u = col - NSC; }
    else                      { base = 1024; u = col - NSC - MUL; }
    Bt1[t] = (_Float16)(Wf2a[j * 1280 + base + u * 8 + c] * (0.1f * RS8));
  } else if (idx < R2) {                        // edge-pass-2 MLP weights (K pad 128)
    int t = idx - R1, col = t >> 7, k = t & 127;
    Bt2[t] = (k < 100) ? (_Float16)(Wf2b[k * NSC + col] * 0.1f) : (_Float16)0.f;
  } else if (idx < R3) {                        // s = [a0|x|0] @ Bs
    int t = idx - R2, col = t >> 7, k = t & 127;
    float v = 0.f;
    if (k < 96)       v = lin2a0[k * NSC + col] * RS96;
    else if (k < 104) v = sc1[(k - 96) * NSC + col] * RS8;
    Bs[t] = (_Float16)v;
  } else {                                      // five 32x32 linears, transposed
    int t = idx - R3, m = t >> 10, r = t & 1023, v = r >> 5, u = r & 31;
    const float* src = (m == 0) ? lin2a1 : (m == 1) ? lin2a2
                     : (m == 2) ? lin1b0 : (m == 3) ? lin1b1 : lin1b2;
    BtLin[t] = (_Float16)(src[u * 32 + v] * RS32);
  }
}

// ---------------------------------------------------------------------------
// Shared geometry: lanes 0..15 compute edge vec / Y / emb for this wave's node
// ---------------------------------------------------------------------------
__device__ __forceinline__ void edge_geom(const int* es, const float* pos, int node,
                                          int w, int lane,
                                          int (&srcL)[8][16], float (&yL)[8][16][8],
                                          float (&embL)[8][16]) {
  if (lane < 16) {
    int e = node * DEG + lane;
    int s = es[e];
    float vx = pos[s * 3 + 0] - pos[node * 3 + 0];
    float vy = pos[s * 3 + 1] - pos[node * 3 + 1];
    float vz = pos[s * 3 + 2] - pos[node * 3 + 2];
    float d  = sqrtf(vx * vx + vy * vy + vz * vz);
    float in = 1.f / (d + 1e-12f);
    float nx = vx * in, ny = vy * in, nz = vz * in;
    yL[w][lane][0] = 1.7320508f * nx;
    yL[w][lane][1] = 1.7320508f * ny;
    yL[w][lane][2] = 1.7320508f * nz;
    yL[w][lane][3] = 3.8729833f * nx * ny;
    yL[w][lane][4] = 3.8729833f * ny * nz;
    yL[w][lane][5] = 1.1180340f * (3.f * nz * nz - 1.f);
    yL[w][lane][6] = 3.8729833f * nx * nz;
    yL[w][lane][7] = 1.9364917f * (nx * nx - ny * ny);
    float diff = (d - 0.9f) / 0.9f;
    embL[w][lane] = (diff < 1.f && diff > -1.f) ? __cosf(1.5707963f * diff) : 0.f;
    srcL[w][lane] = s;
  }
}

// ---------------------------------------------------------------------------
// Kernel 1: edge pass 1.  Per wave = one node (16 edges).
//   z[16 x 800] @ Bt1[800 x 160] (WMMA f16/f32), split into two 5-tile passes
//   (acc[5] = 40 VGPRs -> no spills).  B K-panels are staged into a double-
//   buffered LDS panel with CDNA5 async global->LDS loads; synchronized with
//   s_wait_asynccnt + block barriers.  Then Y-weighted edge reduction.
// ---------------------------------------------------------------------------
__global__ void k_edge1(const int* __restrict__ es, const float* __restrict__ pos,
                        const float* __restrict__ W1a, const _Float16* __restrict__ xs_h,
                        const _Float16* __restrict__ Bt1,
                        _Float16* __restrict__ a0h, _Float16* __restrict__ a1h,
                        _Float16* __restrict__ a2h) {
  __shared__ _Float16 haL[8][16][104];
  __shared__ int      srcL[8][16];
  __shared__ float    yL[8][16][8];
  __shared__ float    embL[8][16];
  __shared__ _Float16 panel[2][80][32];   // 5-tile K-panel, double buffered (10 KB)

  int tid = threadIdx.x, w = tid >> 5, lane = tid & 31;
  int g = lane >> 4, row = lane & 15;
  int node = blockIdx.x * 8 + w;

  edge_geom(es, pos, node, w, lane, srcL, yL, embL);
  __syncthreads();
  for (int i = lane; i < 1600; i += 32) {       // ha = silu(emb * W_fc1a)
    int j = i >> 4, r2 = i & 15;
    float v = embL[w][r2] * W1a[j];
    haL[w][r2][j] = (_Float16)f_silu(v);
  }
  __syncthreads();

  int src = srcL[w][row];
  v8h xev = *(const v8h*)(xs_h + src * 8);       // xe row, lives in registers

  unsigned long long btBase = (unsigned long long)(unsigned long long)(uintptr_t)Bt1;
  // Each thread stages 2 x b128 chunks per panel (320 chunks total, constant
  // batch size so the asynccnt thresholds are immediates).
  int c1 = tid;
  int c2 = (tid < 64) ? 256 + tid : 319;

#pragma unroll 1
  for (int pass = 0; pass < 2; ++pass) {
    int colBase = pass * 80;
    v8f acc[5] = {};

    // prologue: stage K-step 0 into buffer 0
    {
      unsigned gv1 = (unsigned)(((colBase + (c1 >> 2)) * KTP + (c1 & 3) * 8) * 2);
      unsigned gv2 = (unsigned)(((colBase + (c2 >> 2)) * KTP + (c2 & 3) * 8) * 2);
      async_ld_b128((unsigned)(uintptr_t)&panel[0][c1 >> 2][(c1 & 3) * 8], gv1, btBase);
      async_ld_b128((unsigned)(uintptr_t)&panel[0][c2 >> 2][(c2 & 3) * 8], gv2, btBase);
    }

#pragma unroll 1
    for (int ks = 0; ks < 25; ++ks) {
      int buf = ks & 1;
      if (ks + 1 < 25) {                         // stage next panel, then drain cur
        int K0n = (ks + 1) * 32;
        unsigned gv1 = (unsigned)(((colBase + (c1 >> 2)) * KTP + K0n + (c1 & 3) * 8) * 2);
        unsigned gv2 = (unsigned)(((colBase + (c2 >> 2)) * KTP + K0n + (c2 & 3) * 8) * 2);
        async_ld_b128((unsigned)(uintptr_t)&panel[buf ^ 1][c1 >> 2][(c1 & 3) * 8], gv1, btBase);
        async_ld_b128((unsigned)(uintptr_t)&panel[buf ^ 1][c2 >> 2][(c2 & 3) * 8], gv2, btBase);
        asm volatile("s_wait_asynccnt 0x2" ::: "memory");
      } else {
        asm volatile("s_wait_asynccnt 0x0" ::: "memory");
      }
      __syncthreads();                           // panel[buf] visible to all waves

      int j0 = ks * 4;                           // (ks*32)/8
      _Float16 hA = haL[w][row][j0 + g];
      _Float16 hB = haL[w][row][j0 + 2 + g];
      v16h af;
#pragma unroll
      for (int i = 0; i < 16; ++i) af[i] = ((i < 8) ? hA : hB) * xev[i & 7];
#pragma unroll
      for (int t = 0; t < 5; ++t) {
        const _Float16* bp = &panel[buf][t * 16 + row][g * 8];
        v16h bf = mk16(*(const v8h*)bp, *(const v8h*)(bp + 16));
        acc[t] = wmma16(af, bf, acc[t]);
      }
      __syncthreads();                           // done reading before buf reuse
    }

    // edge->node reduction weighted by Y (D layout: lane=col, 8 rows/lane)
#pragma unroll
    for (int t = 0; t < 5; ++t) {
      int gt = colBase / 16 + t;                 // global tile index 0..9
      if (gt < 6) {                              // a0 block (Y0 = 1)
        float s = 0.f;
#pragma unroll
        for (int r = 0; r < 8; ++r) s += acc[t][r];
        s += __shfl_xor(s, 16, 32);
        if (lane < 16) a0h[node * 96 + gt * 16 + lane] = (_Float16)(s * RS16);
      } else if (gt < 8) {                       // a1 blocks (x Y1[m])
#pragma unroll
        for (int m = 0; m < 3; ++m) {
          float s = 0.f;
#pragma unroll
          for (int r = 0; r < 8; ++r) s += acc[t][r] * yL[w][r + 8 * g][m];
          s += __shfl_xor(s, 16, 32);
          if (lane < 16)
            a1h[node * 96 + m * 32 + (gt - 6) * 16 + lane] = (_Float16)(s * RS16);
        }
      } else {                                   // a2 blocks (x Y2[m])
#pragma unroll
        for (int m = 0; m < 5; ++m) {
          float s = 0.f;
#pragma unroll
          for (int r = 0; r < 8; ++r) s += acc[t][r] * yL[w][r + 8 * g][3 + m];
          s += __shfl_xor(s, 16, 32);
          if (lane < 16)
            a2h[node * 160 + m * 32 + (gt - 8) * 16 + lane] = (_Float16)(s * RS16);
        }
      }
    }
  }
}

// ---------------------------------------------------------------------------
// Kernel 2: s = [a0|x|0] @ Bs ;  h0 = silu(s[:, :32]) ; g = sigmoid(s[:, 32:96])
// ---------------------------------------------------------------------------
__global__ void k_s(const _Float16* __restrict__ a0h, const _Float16* __restrict__ xh,
                    const _Float16* __restrict__ Bs,
                    _Float16* __restrict__ h0h, _Float16* __restrict__ gh) {
  int tid = threadIdx.x, w = tid >> 5, lane = tid & 31;
  int g = lane >> 4, rlo = lane & 15;
  int row0 = blockIdx.x * 128 + w * 16;
  int arow = row0 + rlo;
  v8f acc[6] = {};
#pragma unroll
  for (int ks = 0; ks < 4; ++ks) {
    int K0 = ks * 32;
    v8h c0, c1;
    if (ks < 3) {
      const _Float16* ap = a0h + arow * 96;
      c0 = *(const v8h*)(ap + K0 + g * 8);
      c1 = *(const v8h*)(ap + K0 + 16 + g * 8);
    } else {
      v8h z = {};
      c0 = (g == 0) ? *(const v8h*)(xh + arow * 8) : z;
      c1 = z;
    }
    v16h af = mk16(c0, c1);
#pragma unroll
    for (int t = 0; t < 6; ++t) {
      const _Float16* bp = Bs + (t * 16 + rlo) * 128 + K0 + g * 8;
      v16h bf = mk16(*(const v8h*)bp, *(const v8h*)(bp + 16));
      acc[t] = wmma16(af, bf, acc[t]);
    }
  }
#pragma unroll
  for (int t = 0; t < 6; ++t) {
    int col = t * 16 + rlo;
#pragma unroll
    for (int r = 0; r < 8; ++r) {
      int M = row0 + r + 8 * g;
      float v = acc[t][r];
      if (t < 2) h0h[M * 32 + col]        = (_Float16)f_silu(v);
      else       gh[M * 64 + (col - 32)]  = (_Float16)f_sigmoid(v);
    }
  }
}

// ---------------------------------------------------------------------------
// Kernel 3: generic rows x 32 @ 32 x 32 WMMA with optional sigmoid-gate epilogue
// ---------------------------------------------------------------------------
__global__ void k_gemm32(const _Float16* __restrict__ A, const _Float16* __restrict__ Bt,
                         _Float16* __restrict__ out, const _Float16* __restrict__ gate,
                         int gateOff, int rpn) {
  int tid = threadIdx.x, w = tid >> 5, lane = tid & 31;
  int g = lane >> 4, rlo = lane & 15;
  int row0 = blockIdx.x * 128 + w * 16;
  int arow = row0 + rlo;
  const _Float16* ap = A + arow * 32;
  v16h af = mk16(*(const v8h*)(ap + g * 8), *(const v8h*)(ap + 16 + g * 8));
  v8f acc[2] = {};
#pragma unroll
  for (int t = 0; t < 2; ++t) {
    const _Float16* bp = Bt + (t * 16 + rlo) * 32 + g * 8;
    v16h bf = mk16(*(const v8h*)bp, *(const v8h*)(bp + 16));
    acc[t] = wmma16(af, bf, acc[t]);
  }
#pragma unroll
  for (int t = 0; t < 2; ++t) {
    int col = t * 16 + rlo;
#pragma unroll
    for (int r = 0; r < 8; ++r) {
      int M = row0 + r + 8 * g;
      float v = acc[t][r];
      if (gate) v *= (float)gate[(M / rpn) * 64 + gateOff + col];
      out[M * 32 + col] = (_Float16)v;
    }
  }
}

// ---------------------------------------------------------------------------
// Kernel 4: edge pass 2.  Per wave = one node.
//   wb = hb[16x128] @ Bt2[128x96] (WMMA), then sum over the whole 16x96 tile of
//   wb ⊙ q (q built from h0l/h1l/h2l[src] and Y) -> per-node message sum.
// ---------------------------------------------------------------------------
__global__ void k_edge2(const int* __restrict__ es, const float* __restrict__ pos,
                        const float* __restrict__ W1b, const _Float16* __restrict__ Bt2,
                        const _Float16* __restrict__ h0l, const _Float16* __restrict__ h1l,
                        const _Float16* __restrict__ h2l, const _Float16* __restrict__ h0h,
                        const float* __restrict__ sc2, const float* __restrict__ lin2b,
                        float* __restrict__ dout) {
  __shared__ _Float16 hbL[8][16][128];
  __shared__ int      srcL[8][16];
  __shared__ float    yL[8][16][8];
  __shared__ float    embL[8][16];
  int tid = threadIdx.x, w = tid >> 5, lane = tid & 31;
  int g = lane >> 4, row = lane & 15;
  int node = blockIdx.x * 8 + w;

  edge_geom(es, pos, node, w, lane, srcL, yL, embL);
  __syncthreads();
  for (int i = lane; i < 2048; i += 32) {        // hb = silu(emb*W_fc1b), K-pad 128
    int j = i >> 4, r2 = i & 15;
    float v = (j < 100) ? f_silu(embL[w][r2] * W1b[j]) : 0.f;
    hbL[w][r2][j] = (_Float16)v;
  }
  __syncthreads();

  v8f acc[6] = {};
#pragma unroll
  for (int ks = 0; ks < 4; ++ks) {
    int K0 = ks * 32;
    v16h af = mk16(*(const v8h*)&hbL[w][row][K0 + g * 8],
                   *(const v8h*)&hbL[w][row][K0 + 16 + g * 8]);
#pragma unroll
    for (int t = 0; t < 6; ++t) {
      const _Float16* bp = Bt2 + (t * 16 + row) * 128 + K0 + g * 8;
      v16h bf = mk16(*(const v8h*)bp, *(const v8h*)(bp + 16));
      acc[t] = wmma16(af, bf, acc[t]);
    }
  }

  float partial = 0.f;
#pragma unroll
  for (int t = 0; t < 6; ++t) {
    int col = t * 16 + row;
#pragma unroll
    for (int r = 0; r < 8; ++r) {
      int er = r + 8 * g;
      int s  = srcL[w][er];
      float q;
      if (t < 2) {
        q = (float)h0l[s * 32 + col];
      } else if (t < 4) {
        int u = col - 32;
        q = RS3 * ((float)h1l[(s * 3 + 0) * 32 + u] * yL[w][er][0] +
                   (float)h1l[(s * 3 + 1) * 32 + u] * yL[w][er][1] +
                   (float)h1l[(s * 3 + 2) * 32 + u] * yL[w][er][2]);
      } else {
        int u = col - 64;
        float a = 0.f;
#pragma unroll
        for (int m = 0; m < 5; ++m)
          a += (float)h2l[(s * 5 + m) * 32 + u] * yL[w][er][3 + m];
        q = RS5 * a;
      }
      partial += acc[t][r] * q;
    }
  }
#pragma unroll
  for (int o = 16; o >= 1; o >>= 1) partial += __shfl_xor(partial, o, 32);

  float hv = (float)h0h[node * 32 + lane] * sc2[lane];
#pragma unroll
  for (int o = 16; o >= 1; o >>= 1) hv += __shfl_xor(hv, o, 32);

  if (lane == 0) {
    float nodeOut = partial * (RS16 * RS96) * lin2b[0] + hv * RS32;
    atomicAdd(&dout[node >> 9], nodeOut * RS512);
  }
}

// ---------------------------------------------------------------------------
// Host launch
// ---------------------------------------------------------------------------
extern "C" void kernel_launch(void* const* d_in, const int* in_sizes, int n_in,
                              void* d_out, int out_size, void* d_ws, size_t ws_size,
                              hipStream_t stream) {
  const float* pos    = (const float*)d_in[0];
  const float* x      = (const float*)d_in[1];
  const int*   esrc   = (const int*)  d_in[3];
  const float* Wf1a   = (const float*)d_in[5];
  const float* Wf2a   = (const float*)d_in[6];
  const float* lin1a  = (const float*)d_in[7];
  const float* lin2a0 = (const float*)d_in[8];
  const float* lin2a1 = (const float*)d_in[9];
  const float* lin2a2 = (const float*)d_in[10];
  const float* sc1    = (const float*)d_in[11];
  const float* lin1b0 = (const float*)d_in[12];
  const float* lin1b1 = (const float*)d_in[13];
  const float* lin1b2 = (const float*)d_in[14];
  const float* Wf1b   = (const float*)d_in[15];
  const float* Wf2b   = (const float*)d_in[16];
  const float* sc2    = (const float*)d_in[17];
  const float* lin2b  = (const float*)d_in[18];
  float* out = (float*)d_out;

  size_t o = 0;
  auto alloc = [&](size_t bytes) { size_t r = o; o = (o + bytes + 255) & ~(size_t)255; return r; };
  char* ws = (char*)d_ws;
  _Float16* xs_h  = (_Float16*)(ws + alloc((size_t)NN * 8 * 2));
  _Float16* xh    = (_Float16*)(ws + alloc((size_t)NN * 8 * 2));
  _Float16* Bt1   = (_Float16*)(ws + alloc((size_t)NOUT * KTP * 2));
  _Float16* Bt2   = (_Float16*)(ws + alloc((size_t)96 * 128 * 2));
  _Float16* Bs    = (_Float16*)(ws + alloc((size_t)96 * 128 * 2));
  _Float16* BtLin = (_Float16*)(ws + alloc((size_t)5 * 1024 * 2));
  _Float16* a0h   = (_Float16*)(ws + alloc((size_t)NN * 96 * 2));
  _Float16* a1h   = (_Float16*)(ws + alloc((size_t)NN * 96 * 2));
  _Float16* a2h   = (_Float16*)(ws + alloc((size_t)NN * 160 * 2));
  _Float16* h0h   = (_Float16*)(ws + alloc((size_t)NN * 32 * 2));
  _Float16* gh    = (_Float16*)(ws + alloc((size_t)NN * 64 * 2));
  _Float16* h1g   = (_Float16*)(ws + alloc((size_t)NN * 96 * 2));
  _Float16* h2g   = (_Float16*)(ws + alloc((size_t)NN * 160 * 2));
  _Float16* h0l   = (_Float16*)(ws + alloc((size_t)NN * 32 * 2));
  _Float16* h1l   = (_Float16*)(ws + alloc((size_t)NN * 96 * 2));
  _Float16* h2l   = (_Float16*)(ws + alloc((size_t)NN * 160 * 2));
  (void)ws_size; (void)n_in; (void)in_sizes; (void)out_size;

  hipMemsetAsync(d_out, 0, (size_t)GG * sizeof(float), stream);

  k_prep<<<(R4 + 255) / 256, 256, 0, stream>>>(x, lin1a, Wf2a, Wf2b, lin2a0, sc1,
                                               lin2a1, lin2a2, lin1b0, lin1b1, lin1b2,
                                               xs_h, xh, Bt1, Bt2, Bs, BtLin);
  k_edge1<<<NN / 8, 256, 0, stream>>>(esrc, pos, Wf1a, xs_h, Bt1, a0h, a1h, a2h);
  k_s<<<NN / 128, 256, 0, stream>>>(a0h, xh, Bs, h0h, gh);
  // h1 = gate(a1 @ lin2a1) ; h2 = gate(a2 @ lin2a2)
  k_gemm32<<<NN * 3 / 128, 256, 0, stream>>>(a1h, BtLin + 0 * 1024, h1g, gh, 0, 3);
  k_gemm32<<<NN * 5 / 128, 256, 0, stream>>>(a2h, BtLin + 1 * 1024, h2g, gh, 32, 5);
  // h0l / h1l / h2l
  k_gemm32<<<NN / 128, 256, 0, stream>>>(h0h, BtLin + 2 * 1024, h0l, nullptr, 0, 1);
  k_gemm32<<<NN * 3 / 128, 256, 0, stream>>>(h1g, BtLin + 3 * 1024, h1l, nullptr, 0, 1);
  k_gemm32<<<NN * 5 / 128, 256, 0, stream>>>(h2g, BtLin + 4 * 1024, h2l, nullptr, 0, 1);
  k_edge2<<<NN / 8, 256, 0, stream>>>(esrc, pos, Wf1b, Bt2, h0l, h1l, h2l, h0h,
                                      sc2, lin2b, out);
}
